// AxialSelfAttention2d_1743756722342
// MI455X (gfx1250) — matrix operand order repaired and checked by reading, not compiled, not run on hardware
//
#include <hip/hip_runtime.h>
#include <hip/hip_bf16.h>

// AxialSelfAttention2d for MI455X (gfx1250, wave32).
// WMMA f16->f32 for all matmuls; GLOBAL_LOAD_ASYNC_TO_LDS_B128 (ASYNCcnt) for
// attention operand staging. B=2, EMBED=512, H=8, Dh=64, S=L=128.

typedef _Float16 half_t;
typedef __attribute__((ext_vector_type(16))) _Float16 v16h;
typedef __attribute__((ext_vector_type(8)))  _Float16 v8h;
typedef __attribute__((ext_vector_type(4)))  _Float16 v4h;
typedef __attribute__((ext_vector_type(4)))  float    v4f;
typedef __attribute__((ext_vector_type(8)))  float    v8f;

#define PLANE 16384       // S*L
#define EMBED 512
#define HPLANE_D 1048576  // 16384 * 64, halfs per (b,h) block in qT/kT/v layouts

// ---- WMMA fragment loaders (CDNA5 wave32 VGPR layouts, CDNA5 ISA) ----
// A 16x32 f16: lane lo holds row m0+lo; halfs [0..7] = K(hi*8 + 0..7),
// halfs [8..15] = K(16 + hi*8 + 0..7). Source stored row-major [m][k], stride ld.
__device__ inline v16h ld_frag_a(const half_t* __restrict__ s, int ld, int m0, int k0, int lane) {
  const int lo = lane & 15, hi = (lane >> 4) & 1;
  const half_t* p = s + (m0 + lo) * ld + k0 + hi * 8;
  v8h x0 = *(const v8h*)(p);
  v8h x1 = *(const v8h*)(p + 16);
  return __builtin_shufflevector(x0, x1, 0,1,2,3,4,5,6,7,8,9,10,11,12,13,14,15);
}
// B 32x16 f16: lane lo holds col n0+lo; halfs [0..7] = K(k0 + hi*8 + 0..7), [8..15] = K(k0 + 16 + hi*8 + 0..7).
// Source stored [n][k] (K contiguous), stride ld.
__device__ inline v16h ld_frag_b(const half_t* __restrict__ s, int ld, int n0, int k0, int lane) {
  const int lo = lane & 15, hi = (lane >> 4) & 1;
  const half_t* p = s + (n0 + lo) * ld + k0 + hi * 8;
  v8h x0 = *(const v8h*)(p);
  v8h x1 = *(const v8h*)(p + 16);
  return __builtin_shufflevector(x0, x1, 0,1,2,3,4,5,6,7,8,9,10,11,12,13,14,15);
}

__device__ inline v8f wmma_f16(v16h a, v16h b, v8f c) {
  return __builtin_amdgcn_wmma_f32_16x16x32_f16(false, a, false, b, (short)0, c, false, false);
}

// Async 16B global->LDS copy (per-lane addresses), tracked by ASYNCcnt.
// CDNA5 ISA: GLOBAL_LOAD_ASYNC_TO_LDS_B128, GV mode.
__device__ inline void async_cp16(uint32_t lds_byte_addr, const half_t* gptr) {
  asm volatile("global_load_async_to_lds_b128 %0, %1, off"
               :: "v"(lds_byte_addr), "v"((unsigned long long)(uintptr_t)gptr)
               : "memory");
}
__device__ inline void async_wait0() {
  asm volatile("s_wait_asynccnt 0x0" ::: "memory");
}

// ============================================================================
// Kernel 1: QKV projection GEMM.  C[o,n] = sum_i W[o,i]*x[b,i,n] + bias[o].
// Epilogue scatters f16 results directly into attention-ready layouts:
//   qT  [b][h][n=s*128+l][d]   (q pre-scaled by Dh^-0.5)
//   kT  [b][h][n][d]
//   vcol[b][h][l][d][s],  vrow[b][h][s][d][l]
// Grid: (N/128=128, O/128=12, B=2), 256 threads (8 waves, 2x4 wave grid).
// ============================================================================
__global__ __launch_bounds__(256) void qkv_gemm_kernel(
    const float* __restrict__ x, const float* __restrict__ W,
    const float* __restrict__ bias, half_t* __restrict__ qT,
    half_t* __restrict__ kT, half_t* __restrict__ vcol, half_t* __restrict__ vrow)
{
  __shared__ half_t As[128 * 32];  // [o][k]
  __shared__ half_t Bs[128 * 32];  // [n][k]

  const int n0 = blockIdx.x * 128;
  const int o0 = blockIdx.y * 128;
  const int bz = blockIdx.z;
  const int tid = threadIdx.x;
  const int lane = tid & 31, wave = tid >> 5;
  const int wm = wave & 1, wn = wave >> 1;   // 2x4 waves -> 64x32 per wave

  v8f acc[4][2];
#pragma unroll
  for (int mt = 0; mt < 4; ++mt)
#pragma unroll
    for (int nt = 0; nt < 2; ++nt) acc[mt][nt] = (v8f){};

  const float* xb = x + (size_t)bz * EMBED * PLANE + n0;

  for (int kk = 0; kk < EMBED; kk += 32) {
    // A tile: 4 float4 loads / thread, v4h stores. Exact trip count (no guards).
#pragma unroll
    for (int it = 0; it < 4; ++it) {
      int i = tid + it * 256;
      int r = i >> 3, c4 = (i & 7) * 4;
      v4f w4 = *(const v4f*)&W[(size_t)(o0 + r) * EMBED + kk + c4];
      v4h h4;
#pragma unroll
      for (int j = 0; j < 4; ++j) h4[j] = (half_t)w4[j];
      *(v4h*)&As[r * 32 + c4] = h4;
    }
    // B tile: float4 along n, transposed 2B stores into [n][k].
#pragma unroll
    for (int it = 0; it < 4; ++it) {
      int i = tid + it * 256;
      int c = i >> 5, n4 = (i & 31) * 4;
      v4f x4 = *(const v4f*)&xb[(size_t)(kk + c) * PLANE + n4];
#pragma unroll
      for (int j = 0; j < 4; ++j) Bs[(n4 + j) * 32 + c] = (half_t)x4[j];
    }
    __syncthreads();

    v16h bfr[2];
#pragma unroll
    for (int nt = 0; nt < 2; ++nt) bfr[nt] = ld_frag_b(Bs, 32, wn * 32 + nt * 16, 0, lane);
#pragma unroll
    for (int mt = 0; mt < 4; ++mt) {
      v16h afr = ld_frag_a(As, 32, wm * 64 + mt * 16, 0, lane);
#pragma unroll
      for (int nt = 0; nt < 2; ++nt)
        acc[mt][nt] = wmma_f16(afr, bfr[nt], acc[mt][nt]);
    }
    __syncthreads();
  }

  // Epilogue. o0 is a multiple of 128 and region boundaries are 512/1024, so
  // the whole block is in exactly one of q/k/v: branch once on a scalar.
  const int lo = lane & 15, hi = (lane >> 4) & 1;
  const int region = o0 >> 9;  // 0=q, 1=k, 2=v (uniform per block)
  const size_t bhbase = (size_t)bz * 8;
  if (region == 0) {
#pragma unroll
    for (int mt = 0; mt < 4; ++mt)
#pragma unroll
      for (int nt = 0; nt < 2; ++nt)
#pragma unroll
        for (int j = 0; j < 8; ++j) {
          int o = o0 + wm * 64 + mt * 16 + j + 8 * hi;
          int n = n0 + wn * 32 + nt * 16 + lo;
          float val = (acc[mt][nt][j] + bias[o]) * 0.125f;  // Dh^-0.5
          int h = (o >> 6) & 7, d = o & 63;
          qT[((bhbase + h) * PLANE + n) * 64 + d] = (half_t)val;
        }
  } else if (region == 1) {
#pragma unroll
    for (int mt = 0; mt < 4; ++mt)
#pragma unroll
      for (int nt = 0; nt < 2; ++nt)
#pragma unroll
        for (int j = 0; j < 8; ++j) {
          int o = o0 + wm * 64 + mt * 16 + j + 8 * hi;
          int n = n0 + wn * 32 + nt * 16 + lo;
          float val = acc[mt][nt][j] + bias[o];
          int h = (o >> 6) & 7, d = o & 63;
          kT[((bhbase + h) * PLANE + n) * 64 + d] = (half_t)val;
        }
  } else {
#pragma unroll
    for (int mt = 0; mt < 4; ++mt)
#pragma unroll
      for (int nt = 0; nt < 2; ++nt)
#pragma unroll
        for (int j = 0; j < 8; ++j) {
          int o = o0 + wm * 64 + mt * 16 + j + 8 * hi;
          int n = n0 + wn * 32 + nt * 16 + lo;
          float val = acc[mt][nt][j] + bias[o];
          int h = (o >> 6) & 7, d = o & 63;
          int s = n >> 7, l = n & 127;
          half_t hv = (half_t)val;
          size_t bh = bhbase + h;
          vcol[((bh * 128 + l) * 64 + d) * 128 + s] = hv;
          vrow[((bh * 128 + s) * 64 + d) * 128 + l] = hv;
        }
  }
}

// ============================================================================
// Kernel 2: axial attention along one axis for one (b, h, fixed-index).
// COL=true : fixed l, positions p = s; writes out.
// COL=false: fixed s, positions p = l; accumulates into out.
// Operands staged into LDS with GLOBAL_LOAD_ASYNC_TO_LDS_B128.
// 8 waves; wave w owns score rows [16w,16w+16). 16 WMMA scores + 16 WMMA PV.
// ============================================================================
template <bool COL>
__global__ __launch_bounds__(256) void axial_attn_kernel(
    const half_t* __restrict__ qT, const half_t* __restrict__ kT,
    const half_t* __restrict__ v, float* __restrict__ out)
{
  __shared__ half_t sm[24576];      // 48 KB
  half_t* sQ = sm;                  // [p][d] 128x64
  half_t* sK = sm + 8192;           // [p][d] 128x64
  half_t* sV = sm + 16384;          // [d][p]  64x128
  half_t* sP = sm;                  // attn [p][t] 128x128 (overlays sQ+sK)

  const int fix = blockIdx.x;       // l (COL) or s (!COL)
  const int h = blockIdx.y, bz = blockIdx.z;
  const int tid = threadIdx.x, lane = tid & 31, wave = tid >> 5;
  const int lo = lane & 15, hi = (lane >> 4) & 1;

  const size_t bh = (size_t)bz * 8 + h;
  const int pstride = COL ? 8192 : 64;                         // halfs between p's
  const size_t gfix = COL ? (size_t)fix * 64 : (size_t)fix * 8192;
  const half_t* qg = qT + bh * HPLANE_D + gfix;
  const half_t* kg = kT + bh * HPLANE_D + gfix;
  const half_t* vg = v  + bh * HPLANE_D + (size_t)fix * 8192;  // contiguous 16KB

  // Async-stage q,k as [p][d] and v as [d][p]: 3 x 1024 16B transfers,
  // exactly 4 iterations per thread (no exec-mask guards).
  const uint32_t smb = (uint32_t)(uintptr_t)sm;
#pragma unroll
  for (int it = 0; it < 4; ++it) {
    int i = tid + it * 256;
    int p = i >> 3, d8 = (i & 7) * 8;
    int go = p * pstride + d8;
    async_cp16(smb + (uint32_t)(i * 8) * 2,         qg + go);
    async_cp16(smb + (uint32_t)(8192 + i * 8) * 2,  kg + go);
    async_cp16(smb + (uint32_t)(16384 + i * 8) * 2, vg + i * 8);
  }
  async_wait0();
  __syncthreads();

  // Scores: S[m,t] = sum_d q[d,m]*k[d,t]; Dh=64 -> 2 ksteps of K=32.
  const int m0 = wave * 16;
  v8f acc[8];
#pragma unroll
  for (int t = 0; t < 8; ++t) acc[t] = (v8f){};
#pragma unroll
  for (int ks = 0; ks < 64; ks += 32) {
    v16h a = ld_frag_a(sQ, 64, m0, ks, lane);
#pragma unroll
    for (int t = 0; t < 8; ++t) {
      v16h bf = ld_frag_b(sK, 64, t * 16, ks, lane);
      acc[t] = wmma_f16(a, bf, acc[t]);
    }
  }

  // Softmax over t: row r = m0 + j + 8*hi lives in one 16-lane group.
#pragma unroll
  for (int j = 0; j < 8; ++j) {
    float m = -3.0e38f;
#pragma unroll
    for (int t = 0; t < 8; ++t) m = fmaxf(m, acc[t][j]);
#pragma unroll
    for (int o2 = 8; o2 > 0; o2 >>= 1) m = fmaxf(m, __shfl_xor(m, o2, 16));
    float s = 0.0f;
#pragma unroll
    for (int t = 0; t < 8; ++t) { float e = __expf(acc[t][j] - m); acc[t][j] = e; s += e; }
#pragma unroll
    for (int o2 = 8; o2 > 0; o2 >>= 1) s += __shfl_xor(s, o2, 16);
    float inv = 1.0f / s;
#pragma unroll
    for (int t = 0; t < 8; ++t) acc[t][j] *= inv;
  }

  __syncthreads();  // all waves done reading sQ/sK before overlay
#pragma unroll
  for (int t = 0; t < 8; ++t)
#pragma unroll
    for (int j = 0; j < 8; ++j) {
      int r = m0 + j + 8 * hi;
      sP[r * 128 + t * 16 + lo] = (half_t)acc[t][j];
    }
  __syncthreads();

  // out2[m,d] = sum_t P[m,t]*v[d,t]; K=128 -> 4 ksteps.
  v8f oacc[4];
#pragma unroll
  for (int dt = 0; dt < 4; ++dt) oacc[dt] = (v8f){};
#pragma unroll
  for (int ks = 0; ks < 128; ks += 32) {
    v16h a = ld_frag_a(sP, 128, m0, ks, lane);
#pragma unroll
    for (int dt = 0; dt < 4; ++dt) {
      v16h bf = ld_frag_b(sV, 128, dt * 16, ks, lane);
      oacc[dt] = wmma_f16(a, bf, oacc[dt]);
    }
  }

  const int sp = COL ? 128 : 1;
  const size_t obase = ((size_t)bz * EMBED + h * 64) * PLANE +
                       (COL ? (size_t)fix : (size_t)fix * 128);
#pragma unroll
  for (int dt = 0; dt < 4; ++dt)
#pragma unroll
    for (int j = 0; j < 8; ++j) {
      int d = dt * 16 + lo;
      int p = m0 + j + 8 * hi;
      size_t oi = obase + (size_t)d * PLANE + (size_t)p * sp;
      float val = oacc[dt][j];
      if (!COL) val += out[oi];   // row pass accumulates onto column pass
      out[oi] = val;
    }
}

extern "C" void kernel_launch(void* const* d_in, const int* in_sizes, int n_in,
                              void* d_out, int out_size, void* d_ws, size_t ws_size,
                              hipStream_t stream) {
  (void)in_sizes; (void)n_in; (void)out_size; (void)ws_size;
  const float* x    = (const float*)d_in[0];  // [2,512,128,128]
  const float* W    = (const float*)d_in[1];  // [1536,512]
  const float* bias = (const float*)d_in[2];  // [1536]
  float* out = (float*)d_out;                 // [2,512,128,128]

  half_t* ws   = (half_t*)d_ws;               // 128 MB of f16 scratch
  half_t* qT   = ws;                          // [2][8][16384][64]
  half_t* kT   = ws + 16777216;
  half_t* vcol = ws + 33554432;               // [2][8][128(l)][64][128(s)]
  half_t* vrow = ws + 50331648;               // [2][8][128(s)][64][128(l)]

  qkv_gemm_kernel<<<dim3(128, 12, 2), 256, 0, stream>>>(x, W, bias, qT, kT, vcol, vrow);
  axial_attn_kernel<true ><<<dim3(128, 8, 2), 256, 0, stream>>>(qT, kT, vcol, out);  // column
  axial_attn_kernel<false><<<dim3(128, 8, 2), 256, 0, stream>>>(qT, kT, vrow, out);  // row (+=)
}
